// VectorQuantizer_63496796504189
// MI455X (gfx1250) — matrix operand order repaired and checked by reading, not compiled, not run on hardware
//
#include <hip/hip_runtime.h>

// ---------------------------------------------------------------------------
// VQ encode (L2 argmin over 8192 codes) + decode, for MI455X (gfx1250).
// GEMM done as split-bf16 (hi+lo) WMMA: dot ~= Ah*Bh + Ah*Bl + Al*Bh.
// argmin uses score = ||e||^2 - 2*dot  (||x||^2 is constant per row).
// 24 WMMAs per v-tile spread over 4 accumulator chains to avoid XDL RAW stalls.
// ---------------------------------------------------------------------------

typedef __bf16        v16bf __attribute__((ext_vector_type(16)));
typedef float         v8f   __attribute__((ext_vector_type(8)));
typedef unsigned int  u32x4 __attribute__((ext_vector_type(4)));
typedef int           vi4g  __attribute__((vector_size(16)));   // matches builtin param

#define VQ_C   256
#define VQ_HW  1024
#define VQ_N   16384
#define VQ_V   8192
#define KSTEPS 8              // 256 / 32
#define NVT    (VQ_V / 16)    // 512 v-tiles of 16 codes
#define ROWS_PER_BLOCK 64
#define MAIN_THREADS   128    // 4 waves, one 16-row tile each
#define LDS_COL_STRIDE 264    // 256 bf16 + 8 pad -> bank spread for ds_load_b128

#if __has_builtin(__builtin_amdgcn_global_load_async_to_lds_b128) && \
    __has_builtin(__builtin_amdgcn_s_wait_asynccnt)
#define VQ_ASYNC 1
#else
#define VQ_ASYNC 0
#endif

__device__ __forceinline__ unsigned short f2bf(float f) {
  unsigned int u = __float_as_uint(f);
  u += 0x7fffu + ((u >> 16) & 1u);       // round-to-nearest-even to bf16
  return (unsigned short)(u >> 16);
}
__device__ __forceinline__ float bf2f(unsigned short h) {
  return __uint_as_float(((unsigned int)h) << 16);
}

union Frag32 {                           // 32 bytes = 16 bf16 = one WMMA A/B frag per lane
  u32x4 q[2];
  v16bf v;
};

// --------------------------- prep: split x ---------------------------------
// x[b][c][h][w] fp32 -> Ahi/Alo[n][c] bf16 with n = b*1024 + h*32 + w
__global__ void vq_prep_x(const float* __restrict__ x,
                          unsigned short* __restrict__ Ahi,
                          unsigned short* __restrict__ Alo) {
  int n = blockIdx.x;                    // 0..16383
  int c = threadIdx.x;                   // 0..255
  int b = n >> 10, hw = n & 1023;
  float v = x[(size_t)(b * VQ_C + c) * VQ_HW + hw];
  unsigned short h = f2bf(v);
  Ahi[(size_t)n * VQ_C + c] = h;
  Alo[(size_t)n * VQ_C + c] = f2bf(v - bf2f(h));
}

// ----------------------- prep: split codebook + ||e||^2 --------------------
__global__ void vq_prep_cb(const float* __restrict__ cb,
                           unsigned short* __restrict__ Bhi,
                           unsigned short* __restrict__ Blo,
                           float* __restrict__ esq) {
  int v = blockIdx.x, c = threadIdx.x;
  float e = cb[(size_t)v * VQ_C + c];
  unsigned short h = f2bf(e);
  Bhi[(size_t)v * VQ_C + c] = h;
  Blo[(size_t)v * VQ_C + c] = f2bf(e - bf2f(h));
  float s = e * e;
#pragma unroll
  for (int off = 16; off > 0; off >>= 1) s += __shfl_down(s, off, 32);
  __shared__ float red[8];
  if ((threadIdx.x & 31) == 0) red[threadIdx.x >> 5] = s;
  __syncthreads();
  if (threadIdx.x == 0) {
    float t = 0.f;
#pragma unroll
    for (int i = 0; i < 8; ++i) t += red[i];
    esq[v] = t;
  }
}

// --------------------------- B-tile staging --------------------------------
// 16 codes x 256 K x bf16 (hi+lo) into LDS; 128 threads * (64B hi + 64B lo).
__device__ __forceinline__ void vq_stage(const unsigned short* __restrict__ Bhi,
                                         const unsigned short* __restrict__ Blo,
                                         unsigned short* ldsHi,
                                         unsigned short* ldsLo,
                                         int vt, int t) {
  int c = t >> 3;                        // 0..15  (code within tile)
  int seg = t & 7;                       // 0..7   (32-K chunk)
  size_t g = (size_t)(vt * 16 + c) * VQ_C + seg * 32;
  int l = c * LDS_COL_STRIDE + seg * 32;
#if VQ_ASYNC
#pragma unroll
  for (int j = 0; j < 4; ++j) {
    __builtin_amdgcn_global_load_async_to_lds_b128(
        (__attribute__((address_space(1))) vi4g*)(Bhi + g + j * 8),
        (__attribute__((address_space(3))) vi4g*)(ldsHi + l + j * 8), 0, 0);
    __builtin_amdgcn_global_load_async_to_lds_b128(
        (__attribute__((address_space(1))) vi4g*)(Blo + g + j * 8),
        (__attribute__((address_space(3))) vi4g*)(ldsLo + l + j * 8), 0, 0);
  }
#else
  const u32x4* gh = (const u32x4*)(Bhi + g);
  const u32x4* gl = (const u32x4*)(Blo + g);
  u32x4* lh = (u32x4*)(ldsHi + l);
  u32x4* ll = (u32x4*)(ldsLo + l);
#pragma unroll
  for (int j = 0; j < 4; ++j) { lh[j] = gh[j]; ll[j] = gl[j]; }
#endif
}

// ------------------------------ main kernel --------------------------------
__global__ __launch_bounds__(MAIN_THREADS) void vq_argmin(
    const unsigned short* __restrict__ Ahi, const unsigned short* __restrict__ Alo,
    const unsigned short* __restrict__ Bhi, const unsigned short* __restrict__ Blo,
    const float* __restrict__ esq, int* __restrict__ out_idx) {

  __shared__ __align__(16) unsigned short lds[2][2][16 * LDS_COL_STRIDE];

  const int t = threadIdx.x;
  const int wave = t >> 5;
  const int lane = t & 31;
  const int m = lane & 15;               // A-row / B-col within tile
  const int half = lane >> 4;            // which 16-lane half
  const int n0 = blockIdx.x * ROWS_PER_BLOCK + wave * 16;

  // Preload this wave's 16x256 A tile (hi+lo) into registers.
  // A layout (matches B): lanes 0-15 hold K [k0..k0+15],
  // lanes 16-31 hold [k0+16..k0+31]; row M = lane%16.
  Frag32 a_hi[KSTEPS], a_lo[KSTEPS];
  {
    const u32x4* ph = (const u32x4*)(Ahi + (size_t)(n0 + m) * VQ_C);
    const u32x4* pl = (const u32x4*)(Alo + (size_t)(n0 + m) * VQ_C);
#pragma unroll
    for (int k = 0; k < KSTEPS; ++k) {
      int e0 = k * 32 + half * 16;       // in bf16 elems; /8 -> 16B chunk index
      a_hi[k].q[0] = ph[e0 >> 3]; a_hi[k].q[1] = ph[(e0 + 8) >> 3];
      a_lo[k].q[0] = pl[e0 >> 3]; a_lo[k].q[1] = pl[(e0 + 8) >> 3];
    }
  }

  vq_stage(Bhi, Blo, &lds[0][0][0], &lds[0][1][0], 0, t);
#if VQ_ASYNC
  __builtin_amdgcn_s_wait_asynccnt(0);
#endif
  __syncthreads();

  float best_s[8];
  int   best_i[8];
#pragma unroll
  for (int i = 0; i < 8; ++i) { best_s[i] = 3.4e38f; best_i[i] = 0; }

  int buf = 0;
  for (int vt = 0; vt < NVT; ++vt) {
    if (vt + 1 < NVT)
      vq_stage(Bhi, Blo, &lds[buf ^ 1][0][0], &lds[buf ^ 1][1][0], vt + 1, t);

    float es = esq[vt * 16 + m];         // hoisted: hide global latency under WMMAs

    v8f acc[4];
#pragma unroll
    for (int a = 0; a < 4; ++a) acc[a] = (v8f){0.f, 0.f, 0.f, 0.f, 0.f, 0.f, 0.f, 0.f};

    const unsigned short* lH = &lds[buf][0][0];
    const unsigned short* lL = &lds[buf][1][0];
    // B (KxN, col-major in LDS): lane holds col m, 16 contiguous K at k0+half*16
    const int lbase = m * LDS_COL_STRIDE + half * 16;
#pragma unroll
    for (int k = 0; k < KSTEPS; ++k) {
      Frag32 bh, bl;
      const u32x4* qh = (const u32x4*)(lH + lbase + k * 32);
      const u32x4* ql = (const u32x4*)(lL + lbase + k * 32);
      bh.q[0] = qh[0]; bh.q[1] = qh[1];
      bl.q[0] = ql[0]; bl.q[1] = ql[1];
      // Round-robin products across 4 accumulator chains (dep distance >= 4).
      acc[(k * 3 + 0) & 3] = __builtin_amdgcn_wmma_f32_16x16x32_bf16(
          false, a_hi[k].v, false, bh.v, (short)0, acc[(k * 3 + 0) & 3], false, false);
      acc[(k * 3 + 1) & 3] = __builtin_amdgcn_wmma_f32_16x16x32_bf16(
          false, a_hi[k].v, false, bl.v, (short)0, acc[(k * 3 + 1) & 3], false, false);
      acc[(k * 3 + 2) & 3] = __builtin_amdgcn_wmma_f32_16x16x32_bf16(
          false, a_lo[k].v, false, bh.v, (short)0, acc[(k * 3 + 2) & 3], false, false);
    }

    // Epilogue: C/D layout — VGPR i, lanes 0-15: row i, N=lane; lanes 16-31: row i+8.
#pragma unroll
    for (int i = 0; i < 8; ++i) {
      float d = es - 2.f * ((acc[0][i] + acc[1][i]) + (acc[2][i] + acc[3][i]));
      if (d < best_s[i]) { best_s[i] = d; best_i[i] = vt * 16 + m; }
    }

#if VQ_ASYNC
    __builtin_amdgcn_s_wait_asynccnt(0);
#endif
    __syncthreads();
    buf ^= 1;
  }

  // Reduce over the 16 columns held by each 16-lane half (xor masks stay in-half).
#pragma unroll
  for (int off = 1; off < 16; off <<= 1) {
#pragma unroll
    for (int i = 0; i < 8; ++i) {
      float os = __shfl_xor(best_s[i], off, 32);
      int   oi = __shfl_xor(best_i[i], off, 32);
      if (os < best_s[i] || (os == best_s[i] && oi < best_i[i])) {
        best_s[i] = os; best_i[i] = oi;
      }
    }
  }
  if (m == 0) {
#pragma unroll
    for (int i = 0; i < 8; ++i) out_idx[n0 + half * 8 + i] = best_i[i];
  }
}

// ------------------------------ decode -------------------------------------
// out[b][c][h][w] = codebook[idx[n]][c], n = b*1024 + hw.  Lanes sweep hw so
// writes are contiguous; codebook gathers hit L2 (8 MB resident).
__global__ void vq_decode(const int* __restrict__ idx, const float* __restrict__ cb,
                          float* __restrict__ out) {
  int lane = threadIdx.x & 31;
  int cg = threadIdx.x >> 5;             // 0..7
  int n = blockIdx.x * 32 + lane;
  int id = idx[n];
  int b = n >> 10, hw = n & 1023;
  int cbase = blockIdx.y * 64 + cg * 8;
#pragma unroll
  for (int j = 0; j < 8; ++j) {
    int c = cbase + j;
    out[(size_t)(b * VQ_C + c) * VQ_HW + hw] = cb[(size_t)id * VQ_C + c];
  }
}

// ------------------------------ launcher -----------------------------------
extern "C" void kernel_launch(void* const* d_in, const int* in_sizes, int n_in,
                              void* d_out, int out_size, void* d_ws, size_t ws_size,
                              hipStream_t stream) {
  (void)in_sizes; (void)n_in; (void)out_size; (void)ws_size;
  const float* x  = (const float*)d_in[0];   // [16,256,32,32] fp32
  const float* cb = (const float*)d_in[1];   // [8192,256] fp32
  float* out = (float*)d_out;                // [16,256,32,32] fp32

  char* ws = (char*)d_ws;
  size_t o = 0;
  auto take = [&](size_t bytes) -> char* {
    char* p = ws + o;
    o += (bytes + 255) & ~(size_t)255;
    return p;
  };
  unsigned short* Ahi = (unsigned short*)take((size_t)VQ_N * VQ_C * 2);  // 8 MB
  unsigned short* Alo = (unsigned short*)take((size_t)VQ_N * VQ_C * 2);  // 8 MB
  unsigned short* Bhi = (unsigned short*)take((size_t)VQ_V * VQ_C * 2);  // 4 MB
  unsigned short* Blo = (unsigned short*)take((size_t)VQ_V * VQ_C * 2);  // 4 MB
  float* esq = (float*)take((size_t)VQ_V * 4);                           // 32 KB
  int*   idx = (int*)take((size_t)VQ_N * 4);                             // 64 KB

  vq_prep_x<<<VQ_N, VQ_C, 0, stream>>>(x, Ahi, Alo);
  vq_prep_cb<<<VQ_V, VQ_C, 0, stream>>>(cb, Bhi, Blo, esq);
  vq_argmin<<<VQ_N / ROWS_PER_BLOCK, MAIN_THREADS, 0, stream>>>(Ahi, Alo, Bhi, Blo, esq, idx);
  vq_decode<<<dim3(VQ_N / 32, VQ_C / 64), 256, 0, stream>>>(idx, cb, out);
}
